// Decoder_2585570312332
// MI455X (gfx1250) — matrix operand order repaired and checked by reading, not compiled, not run on hardware
//
#include <hip/hip_runtime.h>

// ---------------------------------------------------------------------------
// Types for CDNA5 WMMA (wave32): bf16 16x16x32, fp32 accumulate.
// ---------------------------------------------------------------------------
typedef __attribute__((ext_vector_type(16))) __bf16 v16bf;
typedef __attribute__((ext_vector_type(8)))  __bf16 v8bf;
typedef __attribute__((ext_vector_type(8)))  float  v8f;

__device__ inline v16bf mk16(const __bf16* p0, const __bf16* p1) {
  v16bf r;
  v8bf a = *(const v8bf*)p0;
  v8bf b = *(const v8bf*)p1;
#pragma unroll
  for (int i = 0; i < 8; ++i) { r[i] = a[i]; r[i + 8] = b[i]; }
  return r;
}

// ---------------------------------------------------------------------------
// Problem constants
// ---------------------------------------------------------------------------
#define BATCH 512
#define TM1   127      // T-1 steps
#define ENCD  256      // encoder feature dim
#define HID   256      // LSTM hidden
#define NG    1024     // 4*HID gate rows

// LSTM tiling: 32 WGs x 512 thr (16 waves), 16 batch rows per WG (1 M-tile).
// Wave w owns N-tiles [w*4, w*4+4). K split: kt 0..4 register-resident,
// kt 5..7 LDS-resident (192KB slab, loaded once).
#define LWGS   32
#define MB     16
#define GST    1040    // gates LDS row stride (pad vs 64 banks)
#define KREG   5       // K-tiles kept in VGPRs
#define NTW    4       // N-tiles per wave

// ---------------------------------------------------------------------------
// Kernel 0: prep.
//  - w_enc[j] = sum_k W_attn2[0,k] * W_attn1[k, 512+j]   (effective enc weight)
//  - pack W_hh (1024x256 f32) -> bf16 WMMA B-fragments:
//      frag(kt,nt): lane l holds column n = nt*16 + (l&15),
//                   elems e=0..15 -> k = kt*32 + (l<16?0:16) + e
//  - wih_bias: [0..1023]=W_ih, [1024..2047]=b_ih+b_hh
// ---------------------------------------------------------------------------
__global__ void prep_kernel(const float* __restrict__ W_attn1,
                            const float* __restrict__ W_attn2,
                            const float* __restrict__ W_hh,
                            const float* __restrict__ W_ih,
                            const float* __restrict__ b_ih,
                            const float* __restrict__ b_hh,
                            float* __restrict__ w_enc,
                            __bf16* __restrict__ whh_pk,
                            float* __restrict__ wih_bias) {
  int g = blockIdx.x * blockDim.x + threadIdx.x;
  if (g < NG * HID) {                       // 262144 pack elements
    int e  = g & 15;
    int l  = (g >> 4) & 31;
    int tt = g >> 9;                        // kt*64 + nt
    int nt = tt & 63;
    int kt = tt >> 6;
    int n  = nt * 16 + (l & 15);
    int k  = kt * 32 + ((l < 16) ? 0 : 16) + e;
    whh_pk[g] = (__bf16)W_hh[n * HID + k];
  } else if (g < NG * HID + ENCD) {
    int j = g - NG * HID;
    float s = 0.f;
    for (int k = 0; k < 128; ++k) s += W_attn2[k] * W_attn1[k * 768 + 512 + j];
    w_enc[j] = s;
  } else if (g < NG * HID + ENCD + NG) {
    int n = g - (NG * HID + ENCD);
    wih_bias[n] = W_ih[n];
  } else if (g < NG * HID + ENCD + 2 * NG) {
    int n = g - (NG * HID + ENCD + NG);
    wih_bias[NG + n] = b_ih[n] + b_hh[n];
  }
}

// ---------------------------------------------------------------------------
// Kernel A: one workgroup (256 thr) per batch element.
//   stage X_encoded[b] (127x256 f32 = 130KB) in LDS, scores = X . w_enc,
//   softmax over t (shift-invariant => state term dropped exactly),
//   context = beta^T X, then:
//     ytilde[b,t] = context.W_fc[:256] + b_fc + W_fc[256]*y_prev[b,t]
//     out_ctx[b]  = context.W_final[256:512] + b_final
// ---------------------------------------------------------------------------
__global__ void attn_ctx_kernel(const float* __restrict__ Xe,
                                const float* __restrict__ yprev,
                                const float* __restrict__ w_enc,
                                const float* __restrict__ W_fc,
                                const float* __restrict__ b_fc,
                                const float* __restrict__ W_final,
                                const float* __restrict__ b_final,
                                float* __restrict__ ytilde,
                                float* __restrict__ out_ctx) {
  extern __shared__ char smem[];
  float* Xs  = (float*)smem;            // 127*256
  float* we  = Xs + TM1 * ENCD;         // 256
  float* sc  = we + ENCD;               // 128 (scores -> beta)
  float* red = sc + 128;                // scratch

  const int b = blockIdx.x, tid = threadIdx.x;
  const float* Xb = Xe + (size_t)b * TM1 * ENCD;

  for (int i = tid; i < (TM1 * ENCD) / 4; i += 256)
    ((float4*)Xs)[i] = ((const float4*)Xb)[i];
  we[tid] = w_enc[tid];
  __syncthreads();

  if (tid < TM1) {
    float s = 0.f;
    const float* xr = Xs + tid * ENCD;
#pragma unroll 8
    for (int e = 0; e < ENCD; ++e) s += xr[e] * we[e];
    sc[tid] = s;
  }
  __syncthreads();
  if (tid == 0) {
    float mx = -1e30f;
    for (int t = 0; t < TM1; ++t) mx = fmaxf(mx, sc[t]);
    red[0] = mx;
  }
  __syncthreads();
  if (tid < TM1) sc[tid] = __expf(sc[tid] - red[0]);
  __syncthreads();
  if (tid == 0) {
    float s = 0.f;
    for (int t = 0; t < TM1; ++t) s += sc[t];
    red[1] = 1.f / s;
  }
  __syncthreads();
  if (tid < TM1) sc[tid] *= red[1];
  __syncthreads();

  float ctx = 0.f;
#pragma unroll 4
  for (int t = 0; t < TM1; ++t) ctx += sc[t] * Xs[t * ENCD + tid];

  float a1 = ctx * W_fc[tid];           // toward y_tilde
  float a2 = ctx * W_final[ENCD + tid]; // toward final output
  __syncthreads();                      // done reading Xs before reuse
  float* r1 = Xs;
  float* r2 = Xs + 256;
  r1[tid] = a1;
  r2[tid] = a2;
  __syncthreads();
  for (int s2 = 128; s2 > 0; s2 >>= 1) {
    if (tid < s2) { r1[tid] += r1[tid + s2]; r2[tid] += r2[tid + s2]; }
    __syncthreads();
  }
  float cy  = r1[0] + b_fc[0];
  float wfy = W_fc[ENCD];
  if (tid < TM1)
    ytilde[(size_t)b * TM1 + tid] = cy + wfy * yprev[(size_t)b * TM1 + tid];
  if (tid == 0) out_ctx[b] = r2[0] + b_final[0];
}

// ---------------------------------------------------------------------------
// Kernel B: LSTM scan. 32 WGs x 512 thr, 16 batch rows per WG.
// Batch-partitioned recurrence => no inter-WG sync. Per step:
//   gates(16x1024,f32) = d_bf16(16x256) x W_hh^T via v_wmma_f32_16x16x32_bf16
// W_hh residency: kt 0..4 explicitly in VGPRs (160/wave), kt 5..7 in LDS
// (192KB slab) -> zero global traffic in the scan loop.
// ---------------------------------------------------------------------------
__global__ void __launch_bounds__(512)
lstm_scan_kernel(const __bf16* __restrict__ whh_pk,
                 const float* __restrict__ wih_bias,
                 const float* __restrict__ ytilde,
                 const float* __restrict__ out_ctx,
                 const float* __restrict__ W_final,
                 float* __restrict__ out) {
  extern __shared__ char smem[];
  float*  gts     = (float*)smem;            // MB*GST f32        ( 66560 B)
  float*  cst     = gts + MB * GST;          // MB*256 f32        ( 16384 B)
  float*  yt      = cst + MB * HID;          // MB*128 f32        (  8192 B)
  float*  bias    = yt + MB * 128;           // 1024 f32          (  4096 B)
  float*  wih     = bias + NG;               // 1024 f32          (  4096 B)
  float*  wfd     = wih + NG;                // 256 f32           (  1024 B)
  __bf16* dbf     = (__bf16*)(wfd + HID);    // MB*256 bf16       (  8192 B)
  __bf16* whh_lds = dbf + MB * HID;          // 3*64*512 bf16     (196608 B)

  const int tid = threadIdx.x;
  const int b0  = blockIdx.x * MB;

  // ---- init LDS state ----
  for (int i = tid; i < MB * HID; i += 512) { cst[i] = 0.f; dbf[i] = (__bf16)0.f; }
  for (int i = tid; i < NG; i += 512) { wih[i] = wih_bias[i]; bias[i] = wih_bias[NG + i]; }
  for (int i = tid; i < HID; i += 512) wfd[i] = W_final[i];
  for (int i = tid; i < MB * TM1; i += 512) {
    int bb = i / TM1, t = i % TM1;
    yt[bb * 128 + t] = ytilde[(size_t)(b0 + bb) * TM1 + t];
  }
  // copy W_hh fragments kt=5..7 into LDS (contiguous 192KB slab)
  {
    const float4* src = (const float4*)(whh_pk + (size_t)KREG * 64 * 512);
    float4* dst = (float4*)whh_lds;
    for (int i = tid; i < (3 * 64 * 512 * 2) / 16; i += 512) dst[i] = src[i];
  }

  const int wave  = tid >> 5, lane = tid & 31;
  const int nt0   = wave * NTW;                 // 4 N-tiles per wave
  const int arow  = lane & 15;                  // batch row (single M-tile)
  const int akoff = (lane < 16) ? 0 : 16;       // A-layout K base per lane half

  // ---- preload register-resident W_hh fragments (kt 0..KREG-1) ----
  v16bf Breg[NTW][KREG];
#pragma unroll
  for (int j = 0; j < NTW; ++j)
#pragma unroll
    for (int kt = 0; kt < KREG; ++kt) {
      const __bf16* bp = whh_pk + (size_t)(kt * 64 + nt0 + j) * 512 + lane * 16;
      Breg[j][kt] = mk16(bp, bp + 8);
    }
  __syncthreads();

  // ---- sequential scan ----
  for (int t = 0; t < TM1; ++t) {
    v8f acc[NTW] = {};
#pragma unroll
    for (int kt = 0; kt < 8; ++kt) {
      const __bf16* ap = dbf + arow * HID + kt * 32 + akoff;
      v16bf a = mk16(ap, ap + 8);
#pragma unroll
      for (int j = 0; j < NTW; ++j) {
        v16bf bm;
        if (kt < KREG) {
          bm = Breg[j][kt];
        } else {
          const __bf16* bp =
              whh_lds + (size_t)((kt - KREG) * 64 + nt0 + j) * 512 + lane * 16;
          bm = mk16(bp, bp + 8);
        }
        acc[j] = __builtin_amdgcn_wmma_f32_16x16x32_bf16(
            false, a, false, bm, (short)0, acc[j], false, false);
      }
    }
    // scatter C/D tiles (lane = N col, vgpr r = M row, +8 for upper lane half)
    {
      const int ncol0 = (lane & 15);
      const int mrow0 = (lane < 16) ? 0 : 8;
#pragma unroll
      for (int j = 0; j < NTW; ++j) {
        const int ncol = (nt0 + j) * 16 + ncol0;
#pragma unroll
        for (int r = 0; r < 8; ++r)
          gts[(mrow0 + r) * GST + ncol] = acc[j][r];
      }
    }
    __syncthreads();
    // elementwise LSTM update (fp32 c, bf16 d for next GEMM)
    for (int i = tid; i < MB * HID; i += 512) {
      const int bb = i >> 8, h = i & 255;
      const float ytv = yt[bb * 128 + t];
      const float* gr = gts + bb * GST;
      float gi = gr[h]       + ytv * wih[h]       + bias[h];
      float gf = gr[256 + h] + ytv * wih[256 + h] + bias[256 + h];
      float gg = gr[512 + h] + ytv * wih[512 + h] + bias[512 + h];
      float go = gr[768 + h] + ytv * wih[768 + h] + bias[768 + h];
      gi = 1.f / (1.f + __expf(-gi));
      gf = 1.f / (1.f + __expf(-gf));
      go = 1.f / (1.f + __expf(-go));
      gg = tanhf(gg);
      float c = gf * cst[i] + gi * gg;
      cst[i] = c;
      dbf[i] = (__bf16)(go * tanhf(c));
    }
    __syncthreads();
  }

  // y_pred[b] = d_T . W_final[:256] + (context . W_final[256:] + b_final)
  if (tid < MB) {
    float s = 0.f;
#pragma unroll 8
    for (int h = 0; h < HID; ++h) s += (float)dbf[tid * HID + h] * wfd[h];
    out[b0 + tid] = s + out_ctx[b0 + tid];
  }
}

// ---------------------------------------------------------------------------
// Launch. Workspace layout (needs ~800 KB):
//   [0, 524288)        whh_pk  bf16
//   [524288, 525312)   w_enc   f32[256]
//   [525312, 533504)   wih_bias f32[2048]
//   [533504, 793600)   ytilde  f32[512*127]
//   [793600, 795648)   out_ctx f32[512]
// ---------------------------------------------------------------------------
extern "C" void kernel_launch(void* const* d_in, const int* in_sizes, int n_in,
                              void* d_out, int out_size, void* d_ws, size_t ws_size,
                              hipStream_t stream) {
  (void)in_sizes; (void)n_in; (void)out_size; (void)ws_size;
  const float* Xe      = (const float*)d_in[0];
  const float* yprev   = (const float*)d_in[1];
  const float* W_attn1 = (const float*)d_in[2];
  const float* W_attn2 = (const float*)d_in[4];
  const float* W_fc    = (const float*)d_in[6];
  const float* b_fc    = (const float*)d_in[7];
  const float* W_ih    = (const float*)d_in[8];
  const float* W_hh    = (const float*)d_in[9];
  const float* b_ih    = (const float*)d_in[10];
  const float* b_hh    = (const float*)d_in[11];
  const float* W_final = (const float*)d_in[12];
  const float* b_final = (const float*)d_in[13];

  char* ws = (char*)d_ws;
  __bf16* whh_pk  = (__bf16*)ws;
  float* w_enc    = (float*)(ws + 524288);
  float* wih_bias = (float*)(ws + 525312);
  float* ytilde   = (float*)(ws + 533504);
  float* out_ctx  = (float*)(ws + 793600);
  float* out      = (float*)d_out;

  const int prep_items = NG * HID + ENCD + 2 * NG;
  prep_kernel<<<(prep_items + 255) / 256, 256, 0, stream>>>(
      W_attn1, W_attn2, W_hh, W_ih, b_ih, b_hh, w_enc, whh_pk, wih_bias);

  size_t smemA = (size_t)(TM1 * ENCD + ENCD + 128 + 64) * sizeof(float);
  attn_ctx_kernel<<<BATCH, 256, smemA, stream>>>(
      Xe, yprev, w_enc, W_fc, b_fc, W_final, b_final, ytilde, out_ctx);

  size_t smemL = (size_t)MB * GST * 4   // gates
               + (size_t)MB * HID * 4   // c state
               + (size_t)MB * 128 * 4   // y_tilde tile
               + (size_t)NG * 4 * 2     // wih + bias
               + (size_t)HID * 4        // wfd
               + (size_t)MB * HID * 2   // d bf16
               + (size_t)3 * 64 * 512 * 2; // W_hh kt 5..7 slab
  lstm_scan_kernel<<<LWGS, 512, smemL, stream>>>(
      whh_pk, wih_bias, ytilde, out_ctx, W_final, out);
}
